// Block_63058709840611
// MI455X (gfx1250) — compile-verified, matrix operand order checked by reading, not run on hardware
//
#include <hip/hip_runtime.h>

typedef _Float16 half_t;
typedef __attribute__((ext_vector_type(16))) _Float16 v16h;
typedef __attribute__((ext_vector_type(8)))  _Float16 v8h;
typedef __attribute__((ext_vector_type(4)))  _Float16 v4h;
typedef __attribute__((ext_vector_type(8)))  float    v8f;
typedef __attribute__((vector_size(16)))     int      v4i;

#define B_   4
#define T_   2048
#define C_   1024
#define H_   16
#define HS_  64
#define M_   (B_*T_)    // 8192 rows
#define LD3_ (3*C_)     // 3072

#define WMMA_F16(a,b,c) __builtin_amdgcn_wmma_f32_16x16x32_f16(false,(a),false,(b),(short)0,(c),false,false)

#if __has_builtin(__builtin_amdgcn_global_load_async_to_lds_b128)
#define ASYNC_LDS 1
#else
#define ASYNC_LDS 0
#endif

typedef __attribute__((address_space(1))) v4i* gptr_v4i;
typedef __attribute__((address_space(3))) v4i* lptr_v4i;

static __device__ __forceinline__ v16h pack16(v8h lo, v8h hi){
  return __builtin_shufflevector(lo, hi, 0,1,2,3,4,5,6,7,8,9,10,11,12,13,14,15);
}

// 16-byte global -> LDS copy; async (ASYNCcnt) when the builtin exists.
static __device__ __forceinline__ void cp16_g2l(void* l, const void* g){
#if ASYNC_LDS
  __builtin_amdgcn_global_load_async_to_lds_b128(
      (gptr_v4i)(g), (lptr_v4i)(l), 0, 0);
#else
  *(v8h*)l = *(const v8h*)g;
#endif
}

static __device__ __forceinline__ void wait_async(){
#if ASYNC_LDS
  asm volatile("s_wait_asynccnt 0x0" ::: "memory");
#endif
}

// ---------------- LayerNorm: f32 row -> f16 row ----------------
__global__ __launch_bounds__(256) void ln_f32_to_f16(const float* __restrict__ x,
    const float* __restrict__ g, const float* __restrict__ bta,
    half_t* __restrict__ out){
  int row = blockIdx.x, tid = threadIdx.x;
  const float* xr = x + (size_t)row * C_;
  float4 xv = ((const float4*)xr)[tid];
  float xa[4] = {xv.x, xv.y, xv.z, xv.w};
  float s = 0.f, s2 = 0.f;
#pragma unroll
  for (int j = 0; j < 4; j++){ s += xa[j]; s2 += xa[j]*xa[j]; }
#pragma unroll
  for (int m = 16; m >= 1; m >>= 1){ s += __shfl_xor(s, m, 32); s2 += __shfl_xor(s2, m, 32); }
  __shared__ float ws[8], ws2[8];
  __shared__ float smu, srs;
  if ((tid & 31) == 0){ ws[tid >> 5] = s; ws2[tid >> 5] = s2; }
  __syncthreads();
  if (tid == 0){
    float a = 0.f, b2 = 0.f;
#pragma unroll
    for (int i = 0; i < 8; i++){ a += ws[i]; b2 += ws2[i]; }
    float mu = a * (1.f / C_);
    smu = mu;
    srs = rsqrtf(b2 * (1.f / C_) - mu * mu + 1e-5f);
  }
  __syncthreads();
  float mu = smu, rs = srs;
  v4h o;
#pragma unroll
  for (int j = 0; j < 4; j++){
    int c = tid * 4 + j;
    o[j] = (half_t)((xa[j] - mu) * rs * g[c] + bta[c]);
  }
  *(v4h*)(out + (size_t)row * C_ + tid * 4) = o;
}

// ---------------- Weight transpose/convert kernels ----------------
__global__ void wT_f32_to_f16(const float* __restrict__ in, half_t* __restrict__ out,
                              int K, int N){
  size_t idx = (size_t)blockIdx.x * 256 + threadIdx.x;
  if (idx >= (size_t)K * N) return;
  size_t n = idx / K, k = idx % K;
  out[idx] = (half_t)in[k * (size_t)N + n];
}

__global__ void qkvT_f32_to_f16(const float* __restrict__ w, half_t* __restrict__ out){
  size_t idx = (size_t)blockIdx.x * 256 + threadIdx.x;
  if (idx >= (size_t)C_ * C_) return;
  size_t n = idx / C_, k = idx % C_;
  size_t h = n / HS_, s = n % HS_;
  out[idx] = (half_t)w[(h * C_ + k) * HS_ + s];
}

__global__ void build_vT(const half_t* __restrict__ qkv, half_t* __restrict__ vT){
  size_t idx = (size_t)blockIdx.x * 256 + threadIdx.x;
  if (idx >= (size_t)M_ * C_) return;
  size_t tg = idx / C_, col = idx % C_;
  size_t b = tg / T_, t = tg % T_;
  size_t h = col / HS_, s = col % HS_;
  vT[(((b * H_ + h) * HS_ + s) * T_) + t] = qkv[tg * LD3_ + 2 * C_ + col];
}

// ---------------- Register-blocked WMMA GEMM ----------------
// Each wave: 32(M) x 64(N) via 2x4 accumulators; block: 8 waves = 64 x 256.
// flags bit0: store f16, bit1: ReLU. bias/residual nullable.
__global__ __launch_bounds__(256) void wmma_gemm_f16(
    const half_t* __restrict__ A, const half_t* __restrict__ Bt,
    int Mdim, int Ndim, int Kdim,
    const float* __restrict__ bias, const float* __restrict__ residual,
    void* __restrict__ out, int flags){
  int tid  = threadIdx.x;
  int wave = tid >> 5, lane = tid & 31;
  int row  = lane & 15, hid = lane >> 4;
  int m0 = blockIdx.y * 64  + (wave >> 2) * 32;
  int n0 = blockIdx.x * 256 + (wave & 3) * 64;
  const half_t* a0 = A  + (size_t)(m0 + row) * Kdim;
  const half_t* a1 = A  + (size_t)(m0 + 16 + row) * Kdim;
  const half_t* b0 = Bt + (size_t)(n0 + row) * Kdim;
  size_t bstep = (size_t)16 * Kdim;
  v8f acc[2][4] = {};
#pragma unroll 2
  for (int k = 0; k < Kdim; k += 32){
    __builtin_prefetch(a0 + k + 256, 0, 3);
    v16h af[2], bf[4];
    af[0] = pack16(*(const v8h*)(a0 + k + 8 * hid), *(const v8h*)(a0 + k + 16 + 8 * hid));
    af[1] = pack16(*(const v8h*)(a1 + k + 8 * hid), *(const v8h*)(a1 + k + 16 + 8 * hid));
#pragma unroll
    for (int nt = 0; nt < 4; nt++){
      const half_t* bp = b0 + (size_t)nt * bstep + k;
      bf[nt] = pack16(*(const v8h*)(bp + 8 * hid), *(const v8h*)(bp + 16 + 8 * hid));
    }
#pragma unroll
    for (int nt = 0; nt < 4; nt++){
      acc[0][nt] = WMMA_F16(af[0], bf[nt], acc[0][nt]);
      acc[1][nt] = WMMA_F16(af[1], bf[nt], acc[1][nt]);
    }
  }
#pragma unroll
  for (int nt = 0; nt < 4; nt++){
    int n = n0 + nt * 16 + row;
    float bn = bias ? bias[n] : 0.f;
#pragma unroll
    for (int mt = 0; mt < 2; mt++){
#pragma unroll
      for (int r = 0; r < 8; r++){
        int m = m0 + mt * 16 + r + 8 * hid;
        float v = acc[mt][nt][r] + bn;
        if (flags & 2) v = fmaxf(v, 0.f);
        size_t o = (size_t)m * Ndim + n;
        if (residual) v += residual[o];
        if (flags & 1) ((half_t*)out)[o] = (half_t)v;
        else           ((float*)out)[o]  = v;
      }
    }
  }
}

// ---------------- Flash attention ----------------
// One block (8 waves) covers 128 queries of one (b,h); K/V tiles (32 keys)
// are async-staged into LDS once per key block and shared by all 8 waves.
__global__ __launch_bounds__(256) void attention_kernel(
    const half_t* __restrict__ qkv, const half_t* __restrict__ vT,
    half_t* __restrict__ attn_out){
  int tid  = threadIdx.x;
  int wave = tid >> 5, lane = tid & 31;
  int row  = lane & 15, hid = lane >> 4;
  int blk  = blockIdx.x;
  int tile = blk & (T_ / 128 - 1);      // 16 query tiles of 128
  int hh   = (blk >> 4) & (H_ - 1);
  int b    = blk >> 8;
  int q0b  = tile * 128;
  int q0   = q0b + wave * 16;           // this wave's 16 query rows

  __shared__ __align__(16) half_t Kt[32][64];        // [key][s]
  __shared__ __align__(16) half_t Vt[64][32];        // [s][key]
  __shared__ __align__(16) half_t pbuf[8][16][32];   // per-wave P staging
  half_t (*P)[32] = pbuf[wave];

  const half_t* qrow = qkv + ((size_t)(b * T_ + q0 + row)) * LD3_ + hh * HS_;
  v16h qa[2];
#pragma unroll
  for (int ks = 0; ks < 2; ks++){
    v8h lo = *(const v8h*)(qrow + ks * 32 + 8 * hid);
    v8h hi = *(const v8h*)(qrow + ks * 32 + 16 + 8 * hid);
    qa[ks] = pack16(lo, hi);
  }

  float mrun[8], lrun[8];
  v8f accO[4] = {};
#pragma unroll
  for (int r = 0; r < 8; r++){ mrun[r] = -__builtin_inff(); lrun[r] = 0.f; }

  const float scale = 0.125f;           // hs^-0.5
  int nblocks = (q0b + 127) / 32 + 1;   // uniform across the block
  for (int j = 0; j < nblocks; j++){
    int kb0 = j * 32;
    // ---- async stage K (32x64) and V (64x32) tiles into LDS ----
    {
      int kr = tid >> 3, kc = (tid & 7) * 8;     // 256 x 16B chunks
      cp16_g2l(&Kt[kr][kc],
               qkv + (size_t)(b * T_ + kb0 + kr) * LD3_ + C_ + hh * HS_ + kc);
      int vs = tid >> 2, vc = (tid & 3) * 8;
      cp16_g2l(&Vt[vs][vc],
               vT + (size_t)((b * H_ + hh) * HS_ + vs) * T_ + kb0 + vc);
    }
    wait_async();
    __syncthreads();

    // ---- S = Q K^T for 32 keys (two 16-wide n tiles) ----
    v8f s0 = {}, s1 = {};
#pragma unroll
    for (int ks = 0; ks < 2; ks++){
      v8h l0 = *(const v8h*)(&Kt[row][ks * 32 + 8 * hid]);
      v8h h0 = *(const v8h*)(&Kt[row][ks * 32 + 16 + 8 * hid]);
      v8h l1 = *(const v8h*)(&Kt[row + 16][ks * 32 + 8 * hid]);
      v8h h1 = *(const v8h*)(&Kt[row + 16][ks * 32 + 16 + 8 * hid]);
      s0 = WMMA_F16(qa[ks], pack16(l0, h0), s0);
      s1 = WMMA_F16(qa[ks], pack16(l1, h1), s1);
    }
    // ---- scale, causal mask, online softmax ----
#pragma unroll
    for (int r = 0; r < 8; r++){
      int mi  = q0 + r + 8 * hid;
      int n0k = kb0 + row, n1k = n0k + 16;
      float v0 = s0[r] * scale; if (n0k > mi) v0 = -__builtin_inff();
      float v1 = s1[r] * scale; if (n1k > mi) v1 = -__builtin_inff();
      float mx = fmaxf(v0, v1);
#pragma unroll
      for (int msk = 8; msk >= 1; msk >>= 1) mx = fmaxf(mx, __shfl_xor(mx, msk, 32));
      float mnew  = fmaxf(mrun[r], mx);        // finite after block 0
      float alpha = __expf(mrun[r] - mnew);
      mrun[r] = mnew;
      float p0 = __expf(v0 - mnew);
      float p1 = __expf(v1 - mnew);
      float rs = p0 + p1;
#pragma unroll
      for (int msk = 8; msk >= 1; msk >>= 1) rs += __shfl_xor(rs, msk, 32);
      lrun[r] = lrun[r] * alpha + rs;
#pragma unroll
      for (int st = 0; st < 4; st++) accO[st][r] *= alpha;
      P[r + 8 * hid][row]      = (half_t)p0;   // D-layout -> LDS
      P[r + 8 * hid][row + 16] = (half_t)p1;
    }
    asm volatile("s_wait_dscnt 0x0" ::: "memory");
    v8h plo = *(const v8h*)(&P[row][8 * hid]);  // re-read in A-layout
    v8h phi = *(const v8h*)(&P[row][16 + 8 * hid]);
    v16h pa = pack16(plo, phi);
    // ---- O += P V ----
#pragma unroll
    for (int st = 0; st < 4; st++){
      v8h vl = *(const v8h*)(&Vt[st * 16 + row][8 * hid]);
      v8h vh = *(const v8h*)(&Vt[st * 16 + row][16 + 8 * hid]);
      accO[st] = WMMA_F16(pa, pack16(vl, vh), accO[st]);
    }
    __syncthreads();   // protect Kt/Vt before next stage
  }
  // ---- normalize and store f16 ----
#pragma unroll
  for (int st = 0; st < 4; st++){
#pragma unroll
    for (int r = 0; r < 8; r++){
      float v = accO[st][r] / lrun[r];
      size_t o = (size_t)(b * T_ + q0 + r + 8 * hid) * C_ + hh * HS_ + st * 16 + row;
      attn_out[o] = (half_t)v;
    }
  }
}

// ---------------- Host orchestration ----------------
extern "C" void kernel_launch(void* const* d_in, const int* in_sizes, int n_in,
                              void* d_out, int out_size, void* d_ws, size_t ws_size,
                              hipStream_t stream){
  (void)in_sizes; (void)n_in; (void)out_size; (void)ws_size;
  const float* x    = (const float*)d_in[0];
  const float* ln1g = (const float*)d_in[1];
  const float* ln1b = (const float*)d_in[2];
  const float* Wq   = (const float*)d_in[3];
  const float* Wk   = (const float*)d_in[4];
  const float* Wv   = (const float*)d_in[5];
  const float* Wo   = (const float*)d_in[6];
  const float* bo   = (const float*)d_in[7];
  const float* ln2g = (const float*)d_in[8];
  const float* ln2b = (const float*)d_in[9];
  const float* W1   = (const float*)d_in[10];
  const float* b1   = (const float*)d_in[11];
  const float* W2   = (const float*)d_in[12];
  const float* b2   = (const float*)d_in[13];
  float* out = (float*)d_out;

  char* w = (char*)d_ws;
  size_t off = 0;
  auto alloc = [&](size_t bytes)->char*{
    char* p = w + off; off += (bytes + 255) & ~(size_t)255; return p;
  };
  half_t* h16    = (half_t*)alloc((size_t)M_ * C_ * 2);
  half_t* WqkvT  = (half_t*)alloc((size_t)3 * C_ * C_ * 2);
  half_t* qkv16  = (half_t*)alloc((size_t)M_ * LD3_ * 2);
  half_t* vT16   = (half_t*)alloc((size_t)M_ * C_ * 2);
  half_t* attn16 = (half_t*)alloc((size_t)M_ * C_ * 2);
  half_t* WoT    = (half_t*)alloc((size_t)C_ * C_ * 2);
  float*  x1     = (float*) alloc((size_t)M_ * C_ * 4);
  half_t* h2     = (half_t*)alloc((size_t)M_ * C_ * 2);
  half_t* W1T    = (half_t*)alloc((size_t)C_ * 4 * C_ * 2);
  half_t* ff1    = (half_t*)alloc((size_t)M_ * 4 * C_ * 2);
  half_t* W2T    = (half_t*)alloc((size_t)4 * C_ * C_ * 2);

  // LN1(x) -> h (f16)
  ln_f32_to_f16<<<M_, 256, 0, stream>>>(x, ln1g, ln1b, h16);

  // weight convert/transpose to f16 [N x K]
  int nqkv = (C_ * C_ + 255) / 256;
  qkvT_f32_to_f16<<<nqkv, 256, 0, stream>>>(Wq, WqkvT);
  qkvT_f32_to_f16<<<nqkv, 256, 0, stream>>>(Wk, WqkvT + (size_t)C_ * C_);
  qkvT_f32_to_f16<<<nqkv, 256, 0, stream>>>(Wv, WqkvT + (size_t)2 * C_ * C_);
  wT_f32_to_f16<<<nqkv, 256, 0, stream>>>(Wo, WoT, C_, C_);
  int n4 = (4 * C_ * C_ + 255) / 256;
  wT_f32_to_f16<<<n4, 256, 0, stream>>>(W1, W1T, C_, 4 * C_);
  wT_f32_to_f16<<<n4, 256, 0, stream>>>(W2, W2T, 4 * C_, C_);

  // fused QKV projection: [8192 x 1024] x [1024 x 3072] -> qkv f16
  wmma_gemm_f16<<<dim3(LD3_ / 256, M_ / 64), 256, 0, stream>>>(
      h16, WqkvT, M_, LD3_, C_, nullptr, nullptr, qkv16, 1);

  // V transpose for attn@V fragments
  build_vT<<<(M_ * C_) / 256, 256, 0, stream>>>(qkv16, vT16);

  // causal flash attention: one block per (b, h, 128-query tile)
  attention_kernel<<<B_ * H_ * (T_ / 128), 256, 0, stream>>>(qkv16, vT16, attn16);

  // x1 = x + attn @ Wo + bo   (f32)
  wmma_gemm_f16<<<dim3(C_ / 256, M_ / 64), 256, 0, stream>>>(
      attn16, WoT, M_, C_, C_, bo, x, x1, 0);

  // LN2(x1) -> h2 (f16)
  ln_f32_to_f16<<<M_, 256, 0, stream>>>(x1, ln2g, ln2b, h2);

  // ff1 = relu(h2 @ W1 + b1)  (f16)
  wmma_gemm_f16<<<dim3(4 * C_ / 256, M_ / 64), 256, 0, stream>>>(
      h2, W1T, M_, 4 * C_, C_, b1, nullptr, ff1, 3);

  // out = x1 + ff1 @ W2 + b2  (f32)
  wmma_gemm_f16<<<dim3(C_ / 256, M_ / 64), 256, 0, stream>>>(
      ff1, W2T, M_, C_, 4 * C_, b2, x1, out, 0);
}
